// RLSTM_21595095564497
// MI455X (gfx1250) — compile-verified, matrix operand order checked by reading, not run on hardware
//
#include <hip/hip_runtime.h>
#include <stdint.h>

typedef __attribute__((ext_vector_type(16))) __bf16 v16bf;
typedef __attribute__((ext_vector_type(8)))  float  v8f;
typedef __attribute__((ext_vector_type(4)))  float  fx4;   // clang-native float4

#define B_TOT   65536
#define IN_DIM  128
#define H_DIM   256
#define M_DIM   32
#define K_DIM   (IN_DIM + H_DIM)   // 384
#define BM      64                 // rows per block (2 M-tiles per wave)
#define A_STRIDE   (K_DIM + 8)     // 392 bf16 -> conflict-free 16-lane frag reads
#define MED_STRIDE (M_DIM + 4)     // 36 f32   -> conflict-free frag reads

__device__ __forceinline__ float fast_sigmoid(float x) {
    return 1.0f / (1.0f + __expf(-x));
}
__device__ __forceinline__ float fast_tanh(float x) {
    return 1.0f - 2.0f / (__expf(2.0f * x) + 1.0f);
}

__device__ __forceinline__ v8f splat8(float x) {
    v8f r;
    #pragma unroll
    for (int i = 0; i < 8; ++i) r[i] = x;
    return r;
}

__device__ __forceinline__ void store_bf16x4(__bf16* p, fx4 v) {
    union { __bf16 h[4]; uint2 u; } t;
    t.h[0] = (__bf16)v.x; t.h[1] = (__bf16)v.y;
    t.h[2] = (__bf16)v.z; t.h[3] = (__bf16)v.w;
    *(uint2*)p = t.u;   // callers guarantee 8B alignment
}

// Build a 16x32 bf16 A-fragment half-pair from 16 f32 values (ISA layout:
// per lane K = kbase..kbase+7 then kbase+16..kbase+23).
__device__ __forceinline__ v16bf cvt_frag_f32(const float* __restrict__ p) {
    fx4 a = *(const fx4*)(p);
    fx4 b = *(const fx4*)(p + 4);
    fx4 c = *(const fx4*)(p + 16);
    fx4 d = *(const fx4*)(p + 20);
    v16bf r;
    r[0]  = (__bf16)a.x; r[1]  = (__bf16)a.y; r[2]  = (__bf16)a.z; r[3]  = (__bf16)a.w;
    r[4]  = (__bf16)b.x; r[5]  = (__bf16)b.y; r[6]  = (__bf16)b.z; r[7]  = (__bf16)b.w;
    r[8]  = (__bf16)c.x; r[9]  = (__bf16)c.y; r[10] = (__bf16)c.z; r[11] = (__bf16)c.w;
    r[12] = (__bf16)d.x; r[13] = (__bf16)d.y; r[14] = (__bf16)d.z; r[15] = (__bf16)d.w;
    return r;
}

// Async 16B global->LDS copy (CDNA5 async path, tracked by ASYNCcnt).
__device__ __forceinline__ void async_copy_b128(uint32_t lds_addr,
                                                const void* gbase,
                                                uint32_t voff) {
    asm volatile("global_load_async_to_lds_b128 %0, %1, %2"
                 :
                 : "v"(lds_addr), "v"(voff), "s"(gbase)
                 : "memory");
}
__device__ __forceinline__ void wait_async_all() {
    asm volatile("s_wait_asynccnt 0" ::: "memory");
}

// ---------------------------------------------------------------------------
// Prep: convert Wi [256x384] f32 and Wm [256x32] f32 to bf16 in workspace.
// ---------------------------------------------------------------------------
__global__ __launch_bounds__(256) void rlstm_prep(const float* __restrict__ Wi,
                                                  const float* __restrict__ Wm,
                                                  __bf16* __restrict__ ws) {
    const int NWI = H_DIM * K_DIM;   // 98304
    int idx = (blockIdx.x * 256 + threadIdx.x) * 4;
    if (idx < NWI) {
        fx4 v = *(const fx4*)(Wi + idx);
        store_bf16x4(ws + idx, v);
    } else {
        int j = idx - NWI;           // < 8192
        fx4 v = *(const fx4*)(Wm + j);
        store_bf16x4(ws + NWI + j, v);
    }
}

// ---------------------------------------------------------------------------
// Main fused kernel. Block: 256 threads (8 waves), tile = 64 rows x 256 cols.
// Wave w: wcol = w&3 (64-col strip, 4 N-tiles), wrow = w>>2 (32-row half,
// 2 M-tiles). A tile lives in LDS as bf16 (converted at staging); media tile
// staged f32 via async DMA; B fragments stream from L2-resident bf16 Wi.
// ---------------------------------------------------------------------------
__global__ __launch_bounds__(256) void rlstm_main(
    const float*  __restrict__ input, const float* __restrict__ media,
    const float*  __restrict__ h_t,   const float* __restrict__ c_t,
    const float*  __restrict__ bi,    const float* __restrict__ bm,
    const __bf16* __restrict__ wsWi,  const __bf16* __restrict__ wsWm,
    float* __restrict__ out)
{
    __shared__ __bf16 Alds[BM * A_STRIDE];     // 64*392*2 = 50176 B
    __shared__ float  Mlds[BM * MED_STRIDE];   // 64*36*4  =  9216 B

    const int tid     = threadIdx.x;
    const int rowbase = blockIdx.x * BM;

    // ---- media tile: async DMA global->LDS (512 x 16B, 2 per thread)
    #pragma unroll
    for (int i = 0; i < 2; ++i) {
        int idx = i * 256 + tid;               // float4 index
        int r = idx >> 3, k = (idx & 7) * 4;
        uint32_t lds = (uint32_t)(uintptr_t)(&Mlds[r * MED_STRIDE + k]);
        async_copy_b128(lds, media, (uint32_t)((rowbase + r) * M_DIM + k) * 4u);
    }

    // ---- A = [input | h_t]: load f32 (NT), convert to bf16, store to LDS
    #pragma unroll 4
    for (int i = 0; i < (BM * K_DIM) / (256 * 8); ++i) {   // 12 iters, 8 f32 each
        int fidx = (i * 256 + tid) * 8;
        int r = fidx / K_DIM, k = fidx % K_DIM;
        const float* src = (k < IN_DIM)
            ? input + (size_t)(rowbase + r) * IN_DIM + k
            : h_t   + (size_t)(rowbase + r) * H_DIM + (k - IN_DIM);
        fx4 a = __builtin_nontemporal_load((const fx4*)src);
        fx4 b = __builtin_nontemporal_load((const fx4*)(src + 4));
        union { __bf16 h[8]; uint4 q; } t;
        t.h[0] = (__bf16)a.x; t.h[1] = (__bf16)a.y; t.h[2] = (__bf16)a.z; t.h[3] = (__bf16)a.w;
        t.h[4] = (__bf16)b.x; t.h[5] = (__bf16)b.y; t.h[6] = (__bf16)b.z; t.h[7] = (__bf16)b.w;
        *(uint4*)&Alds[r * A_STRIDE + k] = t.q;
    }

    wait_async_all();
    __syncthreads();

    const int wave    = tid >> 5;
    const int lane    = tid & 31;
    const int wcol    = wave & 3;       // 0..3 : 64-col strip
    const int wrow    = wave >> 2;      // 0..1 : 32-row half
    const int lhalf   = lane >> 4;
    const int lmod    = lane & 15;
    const int colbase = wcol * 64;
    const int kbase   = lhalf * 8;      // K base within 32-chunk (ISA layout)

    const __bf16* aptr[2];
    #pragma unroll
    for (int mt = 0; mt < 2; ++mt)
        aptr[mt] = &Alds[(wrow * 32 + mt * 16 + lmod) * A_STRIDE + kbase];

    const __bf16* bptr[4];
    float biv[4];
    #pragma unroll
    for (int nt = 0; nt < 4; ++nt) {
        int col = colbase + nt * 16 + lmod;
        bptr[nt] = wsWi + (size_t)col * K_DIM + kbase;
        biv[nt]  = bi[col];
    }

    // accumulators pre-loaded with bias (folded into WMMA C operand)
    v8f acc[2][4];
    #pragma unroll
    for (int mt = 0; mt < 2; ++mt)
        #pragma unroll
        for (int nt = 0; nt < 4; ++nt)
            acc[mt][nt] = splat8(biv[nt]);

    // ---- main K loop: 384/32 = 12 steps, 8 WMMAs each, no barriers
    for (int kc = 0; kc < K_DIM / 32; ++kc) {
        union { v16bf v; uint4 q[2]; } af[2];
        #pragma unroll
        for (int mt = 0; mt < 2; ++mt) {
            const __bf16* ap = aptr[mt] + kc * 32;
            af[mt].q[0] = *(const uint4*)(ap);
            af[mt].q[1] = *(const uint4*)(ap + 16);
        }
        #pragma unroll
        for (int nt = 0; nt < 4; ++nt) {
            union { v16bf v; uint4 q[2]; } bf;
            const __bf16* bp = bptr[nt] + kc * 32;
            bf.q[0] = *(const uint4*)(bp);
            bf.q[1] = *(const uint4*)(bp + 16);
            #pragma unroll
            for (int mt = 0; mt < 2; ++mt)
                acc[mt][nt] = __builtin_amdgcn_wmma_f32_16x16x32_bf16(
                    false, af[mt].v, false, bf.v, (short)0, acc[mt][nt], false, false);
        }
    }

    // ---- media GEMM (K=32) fused into epilogue + elementwise LSTM math
    const size_t BH = (size_t)B_TOT * H_DIM;
    #pragma unroll
    for (int mt = 0; mt < 2; ++mt) {
        union { v16bf v; uint4 q[2]; } am;
        am.v = cvt_frag_f32(&Mlds[(wrow * 32 + mt * 16 + lmod) * MED_STRIDE + kbase]);

        #pragma unroll
        for (int nt = 0; nt < 4; ++nt) {
            int col = colbase + nt * 16 + lmod;
            union { v16bf v; uint4 q[2]; } bw;
            const __bf16* wp = wsWm + (size_t)col * M_DIM + kbase;
            bw.q[0] = *(const uint4*)(wp);
            bw.q[1] = *(const uint4*)(wp + 16);
            v8f gmed = splat8(bm[col]);        // bias folded into C
            gmed = __builtin_amdgcn_wmma_f32_16x16x32_bf16(
                false, am.v, false, bw.v, (short)0, gmed, false, false);

            #pragma unroll
            for (int r = 0; r < 8; ++r) {
                int row = rowbase + wrow * 32 + mt * 16 + lhalf * 8 + r;
                size_t o = (size_t)row * H_DIM + col;
                float g    = acc[mt][nt][r];
                float gate = fast_sigmoid(g);        // i == f == o
                float tg   = fast_tanh(g);
                float ct   = __builtin_nontemporal_load(&c_t[o]);
                float c    = ct * gate + tg * gate;
                float mg   = fast_sigmoid(gmed[r]);
                float cr   = fast_tanh(c);
                float cf   = c - cr + cr * mg;
                float h    = fast_tanh(cf) * gate;
                __builtin_nontemporal_store(h, &out[o]);
                __builtin_nontemporal_store(h, &out[BH + o]);
                __builtin_nontemporal_store(c, &out[2 * BH + o]);
            }
        }
    }
}

// ---------------------------------------------------------------------------
extern "C" void kernel_launch(void* const* d_in, const int* in_sizes, int n_in,
                              void* d_out, int out_size, void* d_ws, size_t ws_size,
                              hipStream_t stream) {
    const float* input = (const float*)d_in[0];
    const float* media = (const float*)d_in[1];
    const float* h_t   = (const float*)d_in[2];
    const float* c_t   = (const float*)d_in[3];
    const float* Wi    = (const float*)d_in[4];
    const float* bi    = (const float*)d_in[5];
    const float* Wm    = (const float*)d_in[6];
    const float* bm    = (const float*)d_in[7];
    float*  out = (float*)d_out;
    __bf16* ws  = (__bf16*)d_ws;   // Wi bf16 (98304) then Wm bf16 (8192)

    // (98304 + 8192) / 4 = 26624 float4s -> 104 blocks of 256 threads, exact.
    rlstm_prep<<<104, 256, 0, stream>>>(Wi, Wm, ws);

    rlstm_main<<<B_TOT / BM, 256, 0, stream>>>(
        input, media, h_t, c_t, bi, bm,
        ws, ws + H_DIM * K_DIM, out);
}
